// AMNet_ms_6373731467478
// MI455X (gfx1250) — matrix operand order repaired and checked by reading, not compiled, not run on hardware
//
#include <hip/hip_runtime.h>
#include <cstdint>

typedef __attribute__((ext_vector_type(16))) __bf16 v16bf;
typedef __attribute__((ext_vector_type(8)))  float  v8f;
typedef __attribute__((ext_vector_type(4)))  int    v4i_;
typedef __attribute__((address_space(1))) v4i_ gv4i;   // global int4
typedef __attribute__((address_space(3))) v4i_ lv4i;   // LDS int4

#define HID 256
#define INF_ 128
#define FNUM 5
#define KP1 6
#define LDA 40   // LDS row stride in bf16 elems: 32 payload + 8 pad (80B, 16B-aligned, conflict-free)

// Bernstein basis coefficients C[k][i] for degree 5 (x^i coeff of B_{5,k})
__constant__ float BERN_C[KP1][KP1] = {
    {1.f,  -5.f,  10.f, -10.f,   5.f,  -1.f},
    {0.f,   5.f, -20.f,  30.f, -20.f,   5.f},
    {0.f,   0.f,  10.f, -30.f,  30.f, -10.f},
    {0.f,   0.f,   0.f,  10.f, -20.f,  10.f},
    {0.f,   0.f,   0.f,   0.f,   5.f,  -5.f},
    {0.f,   0.f,   0.f,   0.f,   0.f,   1.f}};

__device__ __forceinline__ int imin(int a, int b) { return a < b ? a : b; }
__device__ __forceinline__ float sigmoidf_(float x) { return 1.f / (1.f + __expf(-x)); }

// ---------------------------------------------------------------- async copy (CDNA5)
__device__ __forceinline__ void async_copy16(const __bf16* gsrc, __bf16* ldst) {
#if __has_builtin(__builtin_amdgcn_global_load_async_to_lds_b128)
    __builtin_amdgcn_global_load_async_to_lds_b128((gv4i*)gsrc, (lv4i*)ldst, 0, 0);
#else
    *(uint4*)ldst = *(const uint4*)gsrc;
#endif
}

__device__ __forceinline__ void wait_async_lds() {
#if __has_builtin(__builtin_amdgcn_global_load_async_to_lds_b128) && \
    __has_builtin(__builtin_amdgcn_s_wait_asynccnt)
    __builtin_amdgcn_s_wait_asynccnt(0);
#endif
}

// ---------------------------------------------------------------- utility
__global__ void zero_f32_kernel(float* __restrict__ p, size_t n) {
    size_t i = (size_t)blockIdx.x * blockDim.x + threadIdx.x;
    if (i < n) p[i] = 0.f;
}

__global__ void deg_kernel(const int* __restrict__ src, const int* __restrict__ dst,
                           float* __restrict__ deg, int E) {
    int e = blockIdx.x * blockDim.x + threadIdx.x;
    if (e >= E) return;
    if (src[e] != dst[e]) atomicAdd(&deg[src[e]], 1.0f);
}

__global__ void dis_kernel(const float* __restrict__ deg, float* __restrict__ dis, int n) {
    int i = blockIdx.x * blockDim.x + threadIdx.x;
    if (i >= n) return;
    float d = deg[i];
    dis[i] = (d > 0.f) ? rsqrtf(d) : 0.f;
}

__global__ void normw_kernel(const int* __restrict__ src, const int* __restrict__ dst,
                             const float* __restrict__ dis, float* __restrict__ nw, int E) {
    int e = blockIdx.x * blockDim.x + threadIdx.x;
    if (e >= E) return;
    int s = src[e], d = dst[e];
    nw[e] = (s != d) ? (-0.5f * dis[s] * dis[d]) : 0.f;   // -(dis*ew*dis)/lam_max
}

__global__ void alpha_kernel(const float* __restrict__ fw, float* __restrict__ alpha) {
    int idx = threadIdx.x;
    if (idx >= FNUM * KP1) return;
    int f = idx / KP1, i = idx % KP1;
    float s = 0.f;
    for (int k = 0; k < KP1; ++k) s += sigmoidf_(fw[f * KP1 + k]) * BERN_C[k][i];
    alpha[f * KP1 + i] = s;
}

__global__ void transpose_bf16_kernel(const float* __restrict__ W, __bf16* __restrict__ Wt,
                                      int K, int N) {
    int idx = blockIdx.x * blockDim.x + threadIdx.x;
    if (idx >= K * N) return;
    int k = idx / N, n = idx % N;
    Wt[(size_t)n * K + k] = (__bf16)W[(size_t)k * N + n];
}

// ---------------------------------------------------------------- propagation
__global__ void scatter_kernel(const int* __restrict__ src, const int* __restrict__ dst,
                               const float* __restrict__ nw, const float* __restrict__ z,
                               float* __restrict__ agg, int E) {
    int e = blockIdx.x;
    if (e >= E) return;
    float w = nw[e];
    if (w == 0.f) return;
    int s = src[e], d = dst[e];
    int c = threadIdx.x;
    atomicAdd(&agg[(size_t)d * HID + c], w * z[(size_t)s * HID + c]);
}

__global__ void hf_init_kernel(const float* __restrict__ h, const float* __restrict__ alpha,
                               float* __restrict__ hf, size_t total) {
    size_t idx = (size_t)blockIdx.x * blockDim.x + threadIdx.x;
    if (idx >= total) return;
    float v = h[idx];
    #pragma unroll
    for (int f = 0; f < FNUM; ++f) hf[(size_t)f * total + idx] = alpha[f * KP1] * v;
}

__global__ void combine_kernel(float* __restrict__ znext, const float* __restrict__ zcur,
                               const float* __restrict__ alpha, float* __restrict__ hf,
                               size_t total, int t) {
    size_t idx = (size_t)blockIdx.x * blockDim.x + threadIdx.x;
    if (idx >= total) return;
    float v = znext[idx] + 0.5f * zcur[idx];   // agg + diag_w*z
    znext[idx] = v;
    #pragma unroll
    for (int f = 0; f < FNUM; ++f) hf[(size_t)f * total + idx] += alpha[f * KP1 + t] * v;
}

// ---------------------------------------------------------------- WMMA GEMM
#define WMMA_BF16(a, b, c) \
    __builtin_amdgcn_wmma_f32_16x16x32_bf16(false, (a), false, (b), (short)0, (c), false, false)

__device__ __forceinline__ void load_a_f32(const float* __restrict__ A, int row, int K,
                                           int kk, int hi, v16bf& f) {
    const float* p = A + (size_t)row * K + kk + 8 * hi;   // ISA 16-bit A layout
    #pragma unroll
    for (int j = 0; j < 8; ++j) {
        f[j]     = (__bf16)p[j];
        f[8 + j] = (__bf16)p[16 + j];
    }
}

__device__ __forceinline__ void load_b_bf16(const __bf16* __restrict__ Bt, int nrow, int K,
                                            int kk, int hi, v16bf& f) {
    const __bf16* p = Bt + (size_t)nrow * K + kk + 8 * hi;
    #pragma unroll
    for (int j = 0; j < 8; ++j) { f[j] = p[j]; f[8 + j] = p[16 + j]; }
}

// C[M,N] = act(A[M,K] @ Bt[N,K]^T + bias). Wave = 32x32 tile; block = 8 waves = 64x128.
__global__ void gemm_bf16_kernel(const float* __restrict__ A, const __bf16* __restrict__ Bt,
                                 const float* __restrict__ bias, float* __restrict__ C,
                                 int M, int N, int K, int act) {
    int lane = threadIdx.x & 31;
    int wave = threadIdx.x >> 5;
    int lo = lane & 15, hi = lane >> 4;
    int row_base = blockIdx.y * 64 + (wave >> 2) * 32;
    int col_base = blockIdx.x * 128 + (wave & 3) * 32;
    int r0 = imin(row_base + lo, M - 1);
    int r1 = imin(row_base + 16 + lo, M - 1);
    int n0 = imin(col_base + lo, N - 1);
    int n1 = imin(col_base + 16 + lo, N - 1);

    v8f acc00 = {}, acc01 = {}, acc10 = {}, acc11 = {};
    for (int kk = 0; kk < K; kk += 32) {
        v16bf a0, a1, b0, b1;
        load_a_f32(A, r0, K, kk, hi, a0);
        load_a_f32(A, r1, K, kk, hi, a1);
        load_b_bf16(Bt, n0, K, kk, hi, b0);
        load_b_bf16(Bt, n1, K, kk, hi, b1);
        acc00 = WMMA_BF16(a0, b0, acc00);
        acc01 = WMMA_BF16(a0, b1, acc01);
        acc10 = WMMA_BF16(a1, b0, acc10);
        acc11 = WMMA_BF16(a1, b1, acc11);
    }

    v8f accs[2][2] = {{acc00, acc01}, {acc10, acc11}};
    #pragma unroll
    for (int ti = 0; ti < 2; ++ti) {
        #pragma unroll
        for (int tj = 0; tj < 2; ++tj) {
            int gcol = col_base + tj * 16 + lo;
            if (gcol >= N) continue;
            float bv = bias ? bias[gcol] : 0.f;
            #pragma unroll
            for (int r = 0; r < 8; ++r) {
                int grow = row_base + ti * 16 + r + 8 * hi;
                if (grow < M) {
                    float v = accs[ti][tj][r] + bv;
                    if (act == 1) v = fmaxf(v, 0.f);
                    else if (act == 2) v = tanhf(v);
                    C[(size_t)grow * N + gcol] = v;
                }
            }
        }
    }
}

// ---------------------------------------------------------------- syrk (dominant kernel)
// out = tanh(R @ R^T), R bf16 [Nn, HID]. Block tile 128x128, double-buffered async
// global->LDS staging of 32-wide K slices; wave = 64x32 tile = 8 WMMAs per K-step.
__device__ __forceinline__ void stage_tile(const __bf16* __restrict__ R, int Nn,
                                           int base, int kk, __bf16* tile, int tid) {
    #pragma unroll
    for (int it = 0; it < 2; ++it) {
        int c = tid + it * 256;          // 512 x 16B chunks: 128 rows x 4 segs
        int row = c >> 2, seg = c & 3;
        int gr = imin(base + row, Nn - 1);
        const __bf16* gsrc = R + (size_t)gr * HID + kk + seg * 8;
        __bf16* ldst = tile + row * LDA + seg * 8;
        async_copy16(gsrc, ldst);
    }
}

__global__ void __launch_bounds__(256) syrk_tanh_kernel(const __bf16* __restrict__ R,
                                                        float* __restrict__ Out, int Nn) {
    __shared__ __bf16 As[2][128 * LDA];
    __shared__ __bf16 Bs[2][128 * LDA];
    int tid = threadIdx.x;
    int lane = tid & 31, wave = tid >> 5;
    int lo = lane & 15, hi = lane >> 4;
    int wr = wave >> 2, wc = wave & 3;       // 2x4 wave grid
    int row0 = blockIdx.y * 128, col0 = blockIdx.x * 128;

    v8f acc[4][2] = {};

    stage_tile(R, Nn, row0, 0, As[0], tid);
    stage_tile(R, Nn, col0, 0, Bs[0], tid);
    wait_async_lds();
    __syncthreads();

    for (int kk = 0; kk < HID; kk += 32) {
        int cur = (kk >> 5) & 1;
        int nxt = cur ^ 1;
        if (kk + 32 < HID) {
            stage_tile(R, Nn, row0, kk + 32, As[nxt], tid);
            stage_tile(R, Nn, col0, kk + 32, Bs[nxt], tid);
        }

        v16bf a[4], b[2];
        #pragma unroll
        for (int ti = 0; ti < 4; ++ti) {
            const __bf16* p = &As[cur][(wr * 64 + ti * 16 + lo) * LDA + 8 * hi];
            #pragma unroll
            for (int j = 0; j < 8; ++j) { a[ti][j] = p[j]; a[ti][8 + j] = p[16 + j]; }
        }
        #pragma unroll
        for (int tj = 0; tj < 2; ++tj) {
            const __bf16* p = &Bs[cur][(wc * 32 + tj * 16 + lo) * LDA + 8 * hi];
            #pragma unroll
            for (int j = 0; j < 8; ++j) { b[tj][j] = p[j]; b[tj][8 + j] = p[16 + j]; }
        }
        #pragma unroll
        for (int ti = 0; ti < 4; ++ti) {
            #pragma unroll
            for (int tj = 0; tj < 2; ++tj)
                acc[ti][tj] = WMMA_BF16(a[ti], b[tj], acc[ti][tj]);
        }

        wait_async_lds();
        __syncthreads();
    }

    #pragma unroll
    for (int ti = 0; ti < 4; ++ti) {
        #pragma unroll
        for (int tj = 0; tj < 2; ++tj) {
            int gcol = col0 + wc * 32 + tj * 16 + lo;
            if (gcol >= Nn) continue;
            #pragma unroll
            for (int r = 0; r < 8; ++r) {
                int grow = row0 + wr * 64 + ti * 16 + r + 8 * hi;
                if (grow < Nn)
                    Out[(size_t)grow * Nn + gcol] = tanhf(acc[ti][tj][r]);
            }
        }
    }
}

// ---------------------------------------------------------------- attention
__global__ void rowdot_kernel(const float* __restrict__ a, const float* __restrict__ b,
                              float* __restrict__ logit, int Nn, int f) {
    int wave = threadIdx.x >> 5;
    int lane = threadIdx.x & 31;
    int n = blockIdx.x * 8 + wave;
    if (n >= Nn) return;
    const float* pa = a + (size_t)n * HID;
    const float* pb = b + (size_t)n * HID;
    float acc = 0.f;
    for (int c = lane; c < HID; c += 32) acc += pa[c] * pb[c];
    #pragma unroll
    for (int off = 16; off > 0; off >>= 1) acc += __shfl_xor(acc, off, 32);
    if (lane == 0) logit[(size_t)n * FNUM + f] = acc;
}

__global__ void score_res_kernel(const float* __restrict__ logit, const float* __restrict__ lam,
                                 const float* __restrict__ hf, __bf16* __restrict__ resb,
                                 int Nn) {
    int n = blockIdx.x;
    int c = threadIdx.x;
    float l[FNUM];
    float m = -3.4e38f;
    #pragma unroll
    for (int f = 0; f < FNUM; ++f) { l[f] = logit[(size_t)n * FNUM + f]; m = fmaxf(m, l[f]); }
    float s = 0.f;
    #pragma unroll
    for (int f = 0; f < FNUM; ++f) { l[f] = __expf(l[f] - m); s += l[f]; }
    float inv = 1.f / s;
    size_t total = (size_t)Nn * HID;
    float acc = 0.f;
    #pragma unroll
    for (int f = 0; f < FNUM; ++f) {
        float gate = (f == 0) ? 1.f : sigmoidf_(lam[f]);
        acc += hf[(size_t)f * total + (size_t)n * HID + c] * (l[f] * inv) * gate;
    }
    resb[(size_t)n * HID + c] = (__bf16)acc;
}

// ---------------------------------------------------------------- launch
extern "C" void kernel_launch(void* const* d_in, const int* in_sizes, int n_in,
                              void* d_out, int out_size, void* d_ws, size_t ws_size,
                              hipStream_t stream) {
    const float* x    = (const float*)d_in[0];
    const int*   ei   = (const int*)d_in[1];
    const float* W1   = (const float*)d_in[2];
    const float* b1   = (const float*)d_in[3];
    const float* W2   = (const float*)d_in[4];
    const float* b2   = (const float*)d_in[5];
    const float* fw   = (const float*)d_in[6];
    const float* Wf   = (const float*)d_in[7];
    const float* bf   = (const float*)d_in[8];
    const float* Wx   = (const float*)d_in[9];
    const float* bx   = (const float*)d_in[10];
    const float* lam  = (const float*)d_in[11];
    float* out = (float*)d_out;

    const int Nn = in_sizes[0] / INF_;   // 10000
    const int E  = in_sizes[1] / 2;      // 320000
    const int* src = ei;
    const int* dst = ei + E;
    const size_t total = (size_t)Nn * HID;

    // ---- workspace layout
    char* wsp = (char*)d_ws;
    auto alloc = [&](size_t bytes) -> void* {
        void* p = (void*)wsp;
        wsp += (bytes + 255) & ~(size_t)255;
        return p;
    };
    float*  deg    = (float*)alloc(sizeof(float) * Nn);
    float*  dis    = (float*)alloc(sizeof(float) * Nn);
    float*  nw     = (float*)alloc(sizeof(float) * E);
    float*  alpha  = (float*)alloc(sizeof(float) * FNUM * KP1);
    __bf16* W1t    = (__bf16*)alloc(sizeof(__bf16) * INF_ * HID);
    __bf16* W2t    = (__bf16*)alloc(sizeof(__bf16) * HID * HID);
    __bf16* Wft    = (__bf16*)alloc(sizeof(__bf16) * HID * HID);
    __bf16* Wxt    = (__bf16*)alloc(sizeof(__bf16) * HID * HID);
    float*  T1     = (float*)alloc(sizeof(float) * total);
    float*  h      = (float*)alloc(sizeof(float) * total);
    float*  za     = (float*)alloc(sizeof(float) * total);
    float*  zb     = (float*)alloc(sizeof(float) * total);
    float*  hfbuf  = (float*)alloc(sizeof(float) * FNUM * total);
    float*  tmp    = (float*)alloc(sizeof(float) * total);
    float*  xp     = (float*)alloc(sizeof(float) * total);
    float*  logit  = (float*)alloc(sizeof(float) * Nn * FNUM);
    __bf16* resb   = (__bf16*)alloc(sizeof(__bf16) * total);

    // ---- graph normalization
    zero_f32_kernel<<<(Nn + 255) / 256, 256, 0, stream>>>(deg, (size_t)Nn);
    deg_kernel<<<(E + 255) / 256, 256, 0, stream>>>(src, dst, deg, E);
    dis_kernel<<<(Nn + 255) / 256, 256, 0, stream>>>(deg, dis, Nn);
    normw_kernel<<<(E + 255) / 256, 256, 0, stream>>>(src, dst, dis, nw, E);
    alpha_kernel<<<1, 32, 0, stream>>>(fw, alpha);

    // ---- weight transposes (f32 -> bf16, row = output column)
    transpose_bf16_kernel<<<(INF_ * HID + 255) / 256, 256, 0, stream>>>(W1, W1t, INF_, HID);
    transpose_bf16_kernel<<<(HID * HID + 255) / 256, 256, 0, stream>>>(W2, W2t, HID, HID);
    transpose_bf16_kernel<<<(HID * HID + 255) / 256, 256, 0, stream>>>(Wf, Wft, HID, HID);
    transpose_bf16_kernel<<<(HID * HID + 255) / 256, 256, 0, stream>>>(Wx, Wxt, HID, HID);

    dim3 gemm_block(256);
    dim3 gemm_grid((HID + 127) / 128, (Nn + 63) / 64);

    // ---- input MLP: h = relu(x@W1+b1)@W2 + b2
    gemm_bf16_kernel<<<gemm_grid, gemm_block, 0, stream>>>(x, W1t, b1, T1, Nn, HID, INF_, 1);
    gemm_bf16_kernel<<<gemm_grid, gemm_block, 0, stream>>>(T1, W2t, b2, h, Nn, HID, HID, 0);

    // ---- Bernstein propagation, fused filter accumulation
    size_t tgrid = (total + 255) / 256;
    hf_init_kernel<<<tgrid, 256, 0, stream>>>(h, alpha, hfbuf, total);
    const float* zcur = h;
    float* znext = za;
    float* zother = zb;
    for (int t = 1; t <= KP1 - 1; ++t) {
        zero_f32_kernel<<<tgrid, 256, 0, stream>>>(znext, total);
        scatter_kernel<<<E, HID, 0, stream>>>(src, dst, nw, zcur, znext, E);
        combine_kernel<<<tgrid, 256, 0, stream>>>(znext, zcur, alpha, hfbuf, total, t);
        zcur = znext;
        znext = zother;
        zother = (float*)zcur == za ? zb : za;
        if (t == 1) { znext = zb; zother = za; }
    }

    // ---- attention
    gemm_bf16_kernel<<<gemm_grid, gemm_block, 0, stream>>>(h, Wxt, bx, xp, Nn, HID, HID, 2);
    for (int f = 0; f < FNUM; ++f) {
        gemm_bf16_kernel<<<gemm_grid, gemm_block, 0, stream>>>(
            hfbuf + (size_t)f * total, Wft, bf, tmp, Nn, HID, HID, 2);
        rowdot_kernel<<<(Nn + 7) / 8, 256, 0, stream>>>(tmp, xp, logit, Nn, f);
    }
    score_res_kernel<<<Nn, HID, 0, stream>>>(logit, lam, hfbuf, resb, Nn);

    // ---- out = tanh(res @ res^T)  [N x N]: LDS-staged, double-buffered WMMA syrk
    dim3 syrk_grid((Nn + 127) / 128, (Nn + 127) / 128);
    syrk_tanh_kernel<<<syrk_grid, gemm_block, 0, stream>>>(resb, out, Nn);
}
